// TrSFRNNAgent_84069689852421
// MI455X (gfx1250) — matrix operand order repaired and checked by reading, not compiled, not run on hardware
//
#include <hip/hip_runtime.h>
#include <hip/hip_bf16.h>
#include <math.h>

typedef __attribute__((ext_vector_type(16))) _Float16 v16h;
typedef __attribute__((ext_vector_type(8)))  float    v8f;

#define N_AL    9
#define N_EN    10
#define D_OWN   96
#define D_AL    64
#define D_EN    65
#define AD      32   // attn dim per head
#define NOUT    256  // fused N (128 k/own + 128 v/q)

// ws layout (f16): own-pack [256][96] @0, ally-pack [256][64] @24576, enemy-pack [256][96] @40960
#define WS_OWN_OFF 0
#define WS_AL_OFF  24576
#define WS_EN_OFF  40960
#define WS_TOTAL   65536

// dynamic LDS layout (bytes)
#define LDS_Q_OFF    0        // 16*128 f32  = 8192
#define LDS_A_OFF    8192     // up to 160*96 f16 = 30720
#define LDS_W_OFF    38912    // up to 256*96 f16 = 49152
#define LDS_KV_OFF   88064    // up to 160*256 f16 = 81920
#define LDS_BIAS_OFF 169984   // 768 f32 = 3072 (own|q, ak|av, ek|ev)
#define LDS_BYTES    173056

// ---------------------------------------------------------------------------
// Pack all weight matrices into f16, transposed to [N][K] so B-fragment
// gathers are K-contiguous per lane. Fused N: [0,128)=K-proj, [128,256)=V-proj
// (own stage: [0,128)=Wown, [128,256)=Wq). Enemy K padded 65 -> 96 with zeros.
// ---------------------------------------------------------------------------
__global__ void pack_weights(const float* __restrict__ Wq,  const float* __restrict__ Wak,
                             const float* __restrict__ Wav, const float* __restrict__ Wek,
                             const float* __restrict__ Wev, const float* __restrict__ Wown,
                             _Float16* __restrict__ ws) {
    int idx = blockIdx.x * 256 + threadIdx.x;
    if (idx >= WS_TOTAL) return;
    float v = 0.0f;
    if (idx < WS_AL_OFF) {                       // own: [256][96]
        int n = idx / 96, k = idx % 96;
        v = (n < 128) ? Wown[k * 128 + n] : Wq[k * 128 + (n - 128)];
    } else if (idx < WS_EN_OFF) {                // ally: [256][64]
        int t = idx - WS_AL_OFF;
        int n = t / 64, k = t % 64;
        v = (n < 128) ? Wak[k * 128 + n] : Wav[k * 128 + (n - 128)];
    } else {                                     // enemy: [256][96], K>=65 zero
        int t = idx - WS_EN_OFF;
        int n = t / 96, k = t % 96;
        if (k < D_EN) v = (n < 128) ? Wek[k * 128 + n] : Wev[k * 128 + (n - 128)];
    }
    ws[idx] = (_Float16)v;
}

// ---------------------------------------------------------------------------
// One 16x16 output tile: A[mt] (rows x Kpad f16, LDS) x W^T[nt] ([N][Kpad] f16,
// LDS), kTiles steps of K=32, f32 accumulate via v_wmma_f32_16x16x32_f16.
// Fragments use contiguous k order, k = 16*hi + s, for both A and B (wave32).
// ---------------------------------------------------------------------------
__device__ inline v8f wmma_tile(const _Float16* __restrict__ A, int aStride,
                                const _Float16* __restrict__ Wt, int wStride,
                                int mt, int nt, int kTiles, int lane) {
    v8f acc = {};
    const int l15 = lane & 15;
    const int hi  = (lane >> 4) & 1;
    const int row = mt * 16 + l15;
    const int col = nt * 16 + l15;
    for (int kt = 0; kt < kTiles; ++kt) {
        v16h a, b;
        const int kbA = kt * 32 + hi * 16;
        const int kbB = kt * 32 + hi * 16;
#pragma unroll
        for (int r = 0; r < 8; ++r) {
            const int ka = kbA + (r << 1);
            a[2 * r]     = A[row * aStride + ka];
            a[2 * r + 1] = A[row * aStride + ka + 1];
            const int kb = kbB + (r << 1);
            b[2 * r]     = Wt[col * wStride + kb];
            b[2 * r + 1] = Wt[col * wStride + kb + 1];
        }
        acc = __builtin_amdgcn_wmma_f32_16x16x32_f16(false, a, false, b,
                                                     (short)0, acc, false, false);
    }
    return acc;
}

// ---------------------------------------------------------------------------
// Softmax attention, all 256 threads: 4 lanes per (agent la, head h), each lane
// covers 8 of 32 dims; energies reduced across the 4-lane group via shfl_xor.
// q f32 in LDS, K/V f16 in LDS rows [la*E+e][0..255] (0-127 K, 128-255 V).
// ---------------------------------------------------------------------------
__device__ inline void attention(const float* __restrict__ qlds,
                                 const _Float16* __restrict__ KV,
                                 float* __restrict__ out,
                                 int a0, int E, int outBase, int tid) {
    const int sub  = tid & 3;          // 8-dim slice
    const int task = tid >> 2;         // 0..63
    const int la   = task >> 2;        // agent 0..15
    const int h    = task & 3;         // head
    const int dOff = h * AD + sub * 8;

    const float* qp = qlds + la * 128 + dOff;
    float qv[8];
#pragma unroll
    for (int d = 0; d < 8; ++d) qv[d] = qp[d];

    float en[N_EN];
    float m = -1e30f;
    for (int e = 0; e < E; ++e) {
        const _Float16* kp = KV + (la * E + e) * NOUT + dOff;
        float p = 0.0f;
#pragma unroll
        for (int d = 0; d < 8; ++d) p += qv[d] * (float)kp[d];
        p += __shfl_xor(p, 1, 4);
        p += __shfl_xor(p, 2, 4);
        p *= 0.17677669529663687f;     // 1/sqrt(32)
        en[e] = p;
        m = fmaxf(m, p);
    }
    float sum = 0.0f;
    for (int e = 0; e < E; ++e) { en[e] = __expf(en[e] - m); sum += en[e]; }
    const float inv = 1.0f / sum;

    float acc[8];
#pragma unroll
    for (int d = 0; d < 8; ++d) acc[d] = 0.0f;
    for (int e = 0; e < E; ++e) {
        const _Float16* vp = KV + (la * E + e) * NOUT + 128 + dOff;
        const float w = en[e];
#pragma unroll
        for (int d = 0; d < 8; ++d) acc[d] += w * (float)vp[d];
    }
    float* op = out + (size_t)(a0 + la) * 384 + outBase + dOff;
#pragma unroll
    for (int d = 0; d < 8; ++d) op[d] = acc[d] * inv;
}

// ---------------------------------------------------------------------------
// Fused kernel: 16 agents per workgroup, 256 threads (8 wave32s).
// Stage0: own GEMM  (M=16,K=96,N=256) -> own_feature to global, q to LDS.
// Stage1: ally GEMM (M=144,K=64,N=256) -> K/V f16 in LDS -> attention.
// Stage2: enemy GEMM(M=160,K=96pad,N=256) -> K/V f16 in LDS -> attention.
// ---------------------------------------------------------------------------
__global__ __launch_bounds__(256) void fused_attn(
    const float* __restrict__ own_obs, const float* __restrict__ ally,
    const float* __restrict__ enemy,
    const float* __restrict__ bq,  const float* __restrict__ bak,
    const float* __restrict__ bav, const float* __restrict__ bek,
    const float* __restrict__ bev, const float* __restrict__ bown,
    const _Float16* __restrict__ wpack, float* __restrict__ out) {
    extern __shared__ char smem[];
    float*     qlds  = (float*)(smem + LDS_Q_OFF);
    _Float16*  Alds  = (_Float16*)(smem + LDS_A_OFF);
    _Float16*  Wlds  = (_Float16*)(smem + LDS_W_OFF);
    _Float16*  KV    = (_Float16*)(smem + LDS_KV_OFF);
    float*     blds  = (float*)(smem + LDS_BIAS_OFF);

    const int tid  = threadIdx.x;
    const int lane = tid & 31;
    const int wave = tid >> 5;
    const int l15  = lane & 15;
    const int hi   = (lane >> 4) & 1;
    const int a0   = blockIdx.x * 16;

    // ---- one-time bias preload: [0,256)=bown|bq, [256,512)=bak|bav, [512,768)=bek|bev
    for (int i = tid; i < 768; i += 256) {
        const int j = i & 127;
        float v;
        switch (i >> 7) {
            case 0:  v = bown[j]; break;
            case 1:  v = bq[j];   break;
            case 2:  v = bak[j];  break;
            case 3:  v = bav[j];  break;
            case 4:  v = bek[j];  break;
            default: v = bev[j];  break;
        }
        blds[i] = v;
    }

    // ---------------- Stage 0: own path ----------------
    for (int i = tid; i < 16 * 96; i += 256)
        Alds[i] = (_Float16)own_obs[(size_t)a0 * 96 + i];
    for (int i = tid; i < 256 * 96; i += 256)
        Wlds[i] = wpack[WS_OWN_OFF + i];
    __syncthreads();

    {   // nt = wave in [0,8): cols < 128, own_feature -> global (branch-free)
        const int nt = wave;
        v8f acc = wmma_tile(Alds, 96, Wlds, 96, 0, nt, 3, lane);
        const int col = nt * 16 + l15;
        const float bias = blds[col];
#pragma unroll
        for (int r = 0; r < 8; ++r)
            out[(size_t)(a0 + r + hi * 8) * 384 + col] = acc[r] + bias;
    }
    {   // nt = wave+8 in [8,16): cols >= 128, q -> LDS (branch-free)
        const int nt = wave + 8;
        v8f acc = wmma_tile(Alds, 96, Wlds, 96, 0, nt, 3, lane);
        const int qcol = (nt - 8) * 16 + l15;
        const float bias = blds[128 + qcol];
#pragma unroll
        for (int r = 0; r < 8; ++r)
            qlds[(r + hi * 8) * 128 + qcol] = acc[r] + bias;
    }
    __syncthreads();

    // ---------------- Stage 1: ally path ----------------
    for (int i = tid; i < 144 * 64; i += 256)
        Alds[i] = (_Float16)ally[(size_t)a0 * (N_AL * D_AL) + i];
    for (int i = tid; i < 256 * 64; i += 256)
        Wlds[i] = wpack[WS_AL_OFF + i];
    __syncthreads();

    for (int j = 0; j < 18; ++j) {           // 144 jobs = 9 Mt x 16 Nt
        const int job = wave * 18 + j;
        const int mt = job >> 4, nt = job & 15;
        v8f acc = wmma_tile(Alds, 64, Wlds, 64, mt, nt, 2, lane);
        const int col = nt * 16 + l15;
        const float bias = blds[256 + col];
#pragma unroll
        for (int r = 0; r < 8; ++r) {
            const int row = mt * 16 + r + hi * 8;
            KV[row * NOUT + col] = (_Float16)(acc[r] + bias);
        }
    }
    __syncthreads();
    attention(qlds, KV, out, a0, N_AL, 128, tid);
    __syncthreads();

    // ---------------- Stage 2: enemy path ----------------
    for (int i = tid; i < 160 * 96; i += 256) {
        const int rowi = i / 96, k = i - rowi * 96;
        const float v = (k < D_EN)
            ? enemy[(size_t)a0 * (N_EN * D_EN) + rowi * D_EN + k] : 0.0f;
        Alds[i] = (_Float16)v;
    }
    for (int i = tid; i < 256 * 96; i += 256)
        Wlds[i] = wpack[WS_EN_OFF + i];
    __syncthreads();

    for (int j = 0; j < 20; ++j) {           // 160 jobs = 10 Mt x 16 Nt
        const int job = wave * 20 + j;
        const int mt = job >> 4, nt = job & 15;
        v8f acc = wmma_tile(Alds, 96, Wlds, 96, mt, nt, 3, lane);
        const int col = nt * 16 + l15;
        const float bias = blds[512 + col];
#pragma unroll
        for (int r = 0; r < 8; ++r) {
            const int row = mt * 16 + r + hi * 8;
            KV[row * NOUT + col] = (_Float16)(acc[r] + bias);
        }
    }
    __syncthreads();
    attention(qlds, KV, out, a0, N_EN, 256, tid);
}

extern "C" void kernel_launch(void* const* d_in, const int* in_sizes, int n_in,
                              void* d_out, int out_size, void* d_ws, size_t ws_size,
                              hipStream_t stream) {
    const float* own_obs = (const float*)d_in[0];
    const float* ally    = (const float*)d_in[1];
    const float* enemy   = (const float*)d_in[2];
    const float* Wq   = (const float*)d_in[3];
    const float* bq   = (const float*)d_in[4];
    const float* Wak  = (const float*)d_in[5];
    const float* bak  = (const float*)d_in[6];
    const float* Wav  = (const float*)d_in[7];
    const float* bav  = (const float*)d_in[8];
    const float* Wek  = (const float*)d_in[9];
    const float* bek  = (const float*)d_in[10];
    const float* Wev  = (const float*)d_in[11];
    const float* bev  = (const float*)d_in[12];
    const float* Wown = (const float*)d_in[13];
    const float* bown = (const float*)d_in[14];
    _Float16* wpack = (_Float16*)d_ws;
    float* out = (float*)d_out;

    const int Btot = in_sizes[0] / D_OWN;      // 65536

    (void)hipFuncSetAttribute((const void*)fused_attn,
                              hipFuncAttributeMaxDynamicSharedMemorySize,
                              LDS_BYTES);

    pack_weights<<<(WS_TOTAL + 255) / 256, 256, 0, stream>>>(
        Wq, Wak, Wav, Wek, Wev, Wown, wpack);

    fused_attn<<<Btot / 16, 256, LDS_BYTES, stream>>>(
        own_obs, ally, enemy, bq, bak, bav, bek, bev, bown, wpack, out);
}
